// SpectralAugmentedTransformer_61443802137167
// MI455X (gfx1250) — matrix operand order repaired and checked by reading, not compiled, not run on hardware
//
#include <hip/hip_runtime.h>

// ---------------------------------------------------------------------------
// SpectralAugmentedTransformer for MI455X (gfx1250, wave32, WMMA).
// Math restructure: einsum over i commutes with cumsum over t, so we GEMM the
// raw input x first (g = x @ W per frequency, bf16 WMMA), then do one fused
// register-resident cumsum + modReLU + k-pool pass with lane==frequency
// (K_MAX == 32 == wave width). Bandwidth-bound (~360MB @ 23.3TB/s ~ 15us).
//
// Round-3: real/imag interleaved in one g[b][o][t][k][2] buffer -> packed
// u32 stores in the GEMM epilogue (half the scattered stores) and exactly one
// contiguous 128B line per wave-step in the cumsum kernel.
// ---------------------------------------------------------------------------

typedef __attribute__((ext_vector_type(8)))  __bf16 bf16x8;
typedef __attribute__((ext_vector_type(16))) __bf16 bf16x16;
typedef __attribute__((ext_vector_type(8)))  float  f32x8;

#define SEQ_LEN 2048
#define KMAX    32
#define DSPEC   256
#define DMODEL  768
#define NLAYERS 4
#define BATCH   2
#define MTOT    (BATCH * SEQ_LEN)   // 4096
#define CDIM    (2 * DMODEL)        // 1536

__device__ __forceinline__ float b2f_raw(unsigned int u16bits) {
    unsigned int v = u16bits << 16;
    return __builtin_bit_cast(float, v);
}
__device__ __forceinline__ float b2f(__bf16 b) {
    return b2f_raw((unsigned int)__builtin_bit_cast(unsigned short, b));
}
__device__ __forceinline__ unsigned int f2b_bits(float f) {
    unsigned int v = __builtin_bit_cast(unsigned int, f);
    v += 0x7fffu + ((v >> 16) & 1u);           // round-to-nearest-even
    return v >> 16;
}
__device__ __forceinline__ __bf16 f2b(float f) {
    return __builtin_bit_cast(__bf16, (unsigned short)f2b_bits(f));
}

// A fragment (16x32 bf16, M x K): lane m<16 holds row m, K 0..7 & 16..23;
// lane m+16 holds row m, K 8..15 & 24..31. Two 16B contiguous loads per lane.
__device__ __forceinline__ bf16x16 frag_a(const __bf16* __restrict__ base, int ld, int lane) {
    const int r = lane & 15, h = lane >> 4;
    const __bf16* row = base + (size_t)r * ld;
    bf16x8 lo = *(const bf16x8*)(row + h * 8);
    bf16x8 hi = *(const bf16x8*)(row + 16 + h * 8);
    return __builtin_shufflevector(lo, hi, 0, 1, 2, 3, 4, 5, 6, 7,
                                           8, 9, 10, 11, 12, 13, 14, 15);
}

// B fragment (32x16 bf16, K x N): lane holds one N column; lanes<16 K=0..7 & 16..23,
// lanes>=16 K=8..15 & 24..31. Weights stored [n][k] row-major -> 2x16B loads per lane.
__device__ __forceinline__ bf16x16 frag_b(const __bf16* __restrict__ base, int ld, int lane) {
    const int r = lane & 15, h = lane >> 4;
    const __bf16* row = base + (size_t)r * ld;
    bf16x8 lo = *(const bf16x8*)(row + h * 8);
    bf16x8 hi = *(const bf16x8*)(row + 16 + h * 8);
    return __builtin_shufflevector(lo, hi, 0, 1, 2, 3, 4, 5, 6, 7,
                                           8, 9, 10, 11, 12, 13, 14, 15);
}

#define WMMA(a, b, c) \
    __builtin_amdgcn_wmma_f32_16x16x32_bf16(false, (a), false, (b), (short)0, (c), false, false)

// ---------------------------------------------------------------------------
// Prep kernels
// ---------------------------------------------------------------------------
// conv_w (O,I,K,2) bf16 -> wR[k][o][i], wI[k][o][i]
__global__ __launch_bounds__(256) void k_transpose_w(const __bf16* __restrict__ cw,
                                                     __bf16* __restrict__ wr,
                                                     __bf16* __restrict__ wi) {
    int idx = blockIdx.x * 256 + threadIdx.x;           // k*65536 + o*256 + i
    if (idx >= KMAX * DSPEC * DSPEC) return;
    int k = idx >> 16, o = (idx >> 8) & 255, i = idx & 255;
    size_t src = (((size_t)o * DSPEC + i) * KMAX + k) * 2;
    wr[idx] = cw[src];
    wi[idx] = cw[src + 1];
}

__global__ __launch_bounds__(256) void k_f32_to_bf16(const float* __restrict__ src,
                                                     __bf16* __restrict__ dst, int n) {
    int idx = blockIdx.x * 256 + threadIdx.x;
    if (idx < n) dst[idx] = f2b(src[idx]);
}

// ---------------------------------------------------------------------------
// Kernel 1: per-frequency GEMM  g[b][o][t][k] = {x@wr, x@wi} packed {bf16,bf16}
// One wave computes a 32(M)x32(O) tile for one k. 8 K-steps x 8 WMMA.
// ---------------------------------------------------------------------------
__global__ __launch_bounds__(256) void k_spec_gemm(const __bf16* __restrict__ x,
                                                   const __bf16* __restrict__ wr,
                                                   const __bf16* __restrict__ wi,
                                                   unsigned int* __restrict__ g) {
    int tid = blockIdx.x * 256 + threadIdx.x;
    int wave = tid >> 5, lane = tid & 31;
    if (wave >= KMAX * (MTOT / 32) * (DSPEC / 32)) return;
    int kf   = wave >> 10;            // 128*8 = 1024 tiles per frequency
    int tile = wave & 1023;
    int m0 = (tile >> 3) * 32;
    int n0 = (tile & 7) * 32;
    const __bf16* Wr = wr + (size_t)kf * DSPEC * DSPEC;
    const __bf16* Wi = wi + (size_t)kf * DSPEC * DSPEC;

    f32x8 cr[2][2]{}, ci[2][2]{};
#pragma unroll
    for (int i0 = 0; i0 < DSPEC; i0 += 32) {
        __builtin_prefetch(x + (size_t)m0 * DSPEC + i0 + 32, 0, 0);
        bf16x16 a0  = frag_a(x + (size_t)m0 * DSPEC + i0, DSPEC, lane);
        bf16x16 a1  = frag_a(x + (size_t)(m0 + 16) * DSPEC + i0, DSPEC, lane);
        bf16x16 br0 = frag_b(Wr + (size_t)n0 * DSPEC + i0, DSPEC, lane);
        bf16x16 br1 = frag_b(Wr + (size_t)(n0 + 16) * DSPEC + i0, DSPEC, lane);
        bf16x16 bi0 = frag_b(Wi + (size_t)n0 * DSPEC + i0, DSPEC, lane);
        bf16x16 bi1 = frag_b(Wi + (size_t)(n0 + 16) * DSPEC + i0, DSPEC, lane);
        cr[0][0] = WMMA(a0, br0, cr[0][0]);
        cr[0][1] = WMMA(a0, br1, cr[0][1]);
        cr[1][0] = WMMA(a1, br0, cr[1][0]);
        cr[1][1] = WMMA(a1, br1, cr[1][1]);
        ci[0][0] = WMMA(a0, bi0, ci[0][0]);
        ci[0][1] = WMMA(a0, bi1, ci[0][1]);
        ci[1][0] = WMMA(a1, bi0, ci[1][0]);
        ci[1][1] = WMMA(a1, bi1, ci[1][1]);
    }
    // C/D layout: VGPR j -> M = j (lanes 0-15) or 8+j (lanes 16-31); N = lane&15
    int col_l = lane & 15, rhi = (lane >> 4) * 8;
#pragma unroll
    for (int p = 0; p < 2; ++p)
#pragma unroll
        for (int q = 0; q < 2; ++q) {
            int oc = n0 + q * 16 + col_l;
#pragma unroll
            for (int j = 0; j < 8; ++j) {
                int m = m0 + p * 16 + rhi + j;
                int bb = m >> 11, t = m & (SEQ_LEN - 1);
                size_t gidx = (((size_t)(bb * DSPEC + oc)) * SEQ_LEN + t) * KMAX + kf;
                g[gidx] = f2b_bits(cr[p][q][j]) | (f2b_bits(ci[p][q][j]) << 16);
            }
        }
}

// ---------------------------------------------------------------------------
// Kernel 2: fused cumsum + modReLU + frequency pooling.
// One wave per (b,o); lane == frequency k (wave32 == K_MAX).
// Each step reads exactly one contiguous 128B line of packed {r,i} pairs.
// ---------------------------------------------------------------------------
__global__ __launch_bounds__(256) void k_cumsum_pool(const __bf16* __restrict__ x,
                                                     const unsigned int* __restrict__ g,
                                                     const __bf16* __restrict__ mbias,
                                                     __bf16* __restrict__ pooled) {
    int tid = blockIdx.x * 256 + threadIdx.x;
    int wave = tid >> 5, lane = tid & 31;
    if (wave >= BATCH * DSPEC) return;
    int b = wave >> 8, o = wave & 255;
    const float bias = b2f(mbias[o]);
    const float norm = 0.022097086912079608f;  // 1/sqrt(2048)
    const unsigned int* gp = g + ((size_t)(b * DSPEC + o)) * SEQ_LEN * KMAX;
    float ar = 0.f, ai = 0.f;
    for (int t = 0; t < SEQ_LEN; ++t) {
        unsigned int gv = gp[t * KMAX + lane];
        float gR = b2f_raw(gv & 0xffffu);
        float gI = b2f_raw(gv >> 16);
        float xo = b2f(x[((size_t)(b * SEQ_LEN + t)) * DSPEC + o]);
        // angle = -2*pi*t*k/2048 ; (t*k) mod 2048 keeps v_sin/v_cos in range
        float ph = (float)((t * lane) & (SEQ_LEN - 1)) * (-3.14159265358979f / 1024.f);
        float cw = __cosf(ph), sw = __sinf(ph);
        float Ar = gR + xo;        // identity (+spec) term folded in
        ar += cw * Ar - sw * gI;
        ai += cw * gI + sw * Ar;
        float orr = ar * norm, oii = ai * norm;
        float mag = __fsqrt_rn(orr * orr + oii * oii + 1e-8f);
        float nm  = fmaxf(mag - bias, 0.f);
        float m2  = __fsqrt_rn(nm * nm + 1e-8f);   // phase round-trip cancels
#pragma unroll
        for (int off = 16; off; off >>= 1) m2 += __shfl_xor(m2, off, 32);
        if (lane == 0)
            pooled[((size_t)(b * SEQ_LEN + t)) * DSPEC + o] = f2b(m2 * (1.f / 32.f));
    }
}

// ---------------------------------------------------------------------------
// Kernel 3: hidden = silu(pooled @ w_shared^T + b_shared)   [4096 x 768]
// ---------------------------------------------------------------------------
__global__ __launch_bounds__(256) void k_hidden_gemm(const __bf16* __restrict__ pooled,
                                                     const __bf16* __restrict__ ws,
                                                     const float* __restrict__ bsh,
                                                     __bf16* __restrict__ hidden) {
    int tid = blockIdx.x * 256 + threadIdx.x;
    int wave = tid >> 5, lane = tid & 31;
    if (wave >= (MTOT / 32) * (DMODEL / 32)) return;
    int tm = wave / (DMODEL / 32), tn = wave % (DMODEL / 32);
    int m0 = tm * 32, n0 = tn * 32;
    f32x8 c[2][2]{};
#pragma unroll
    for (int i0 = 0; i0 < DSPEC; i0 += 32) {
        bf16x16 a0 = frag_a(pooled + (size_t)m0 * DSPEC + i0, DSPEC, lane);
        bf16x16 a1 = frag_a(pooled + (size_t)(m0 + 16) * DSPEC + i0, DSPEC, lane);
        bf16x16 b0 = frag_b(ws + (size_t)n0 * DSPEC + i0, DSPEC, lane);
        bf16x16 b1 = frag_b(ws + (size_t)(n0 + 16) * DSPEC + i0, DSPEC, lane);
        c[0][0] = WMMA(a0, b0, c[0][0]);
        c[0][1] = WMMA(a0, b1, c[0][1]);
        c[1][0] = WMMA(a1, b0, c[1][0]);
        c[1][1] = WMMA(a1, b1, c[1][1]);
    }
    int col_l = lane & 15, rhi = (lane >> 4) * 8;
#pragma unroll
    for (int p = 0; p < 2; ++p)
#pragma unroll
        for (int q = 0; q < 2; ++q) {
            int cc = n0 + q * 16 + col_l;
            float bv = bsh[cc];
#pragma unroll
            for (int j = 0; j < 8; ++j) {
                int m = m0 + p * 16 + rhi + j;
                float v = c[p][q][j] + bv;
                float s = v * (1.f / (1.f + __expf(-v)));   // silu
                hidden[(size_t)m * DMODEL + cc] = f2b(s);
            }
        }
}

// ---------------------------------------------------------------------------
// Kernel 4: cond[l] = hidden @ layer_w[l]^T + layer_b[l]   [4 x 4096 x 1536] f32
// ---------------------------------------------------------------------------
__global__ __launch_bounds__(256) void k_cond_gemm(const __bf16* __restrict__ hidden,
                                                   const __bf16* __restrict__ lw,
                                                   const float* __restrict__ lb,
                                                   float* __restrict__ out) {
    int tid = blockIdx.x * 256 + threadIdx.x;
    int wave = tid >> 5, lane = tid & 31;
    const int TILES_L = (MTOT / 32) * (CDIM / 32);  // 128*48 = 6144
    if (wave >= NLAYERS * TILES_L) return;
    int l = wave / TILES_L, rem = wave % TILES_L;
    int tm = rem / (CDIM / 32), tn = rem % (CDIM / 32);
    int m0 = tm * 32, n0 = tn * 32;
    const __bf16* W = lw + (size_t)l * CDIM * DMODEL;
    f32x8 c[2][2]{};
    for (int i0 = 0; i0 < DMODEL; i0 += 32) {
        __builtin_prefetch(hidden + (size_t)m0 * DMODEL + i0 + 32, 0, 0);
        bf16x16 a0 = frag_a(hidden + (size_t)m0 * DMODEL + i0, DMODEL, lane);
        bf16x16 a1 = frag_a(hidden + (size_t)(m0 + 16) * DMODEL + i0, DMODEL, lane);
        bf16x16 b0 = frag_b(W + (size_t)n0 * DMODEL + i0, DMODEL, lane);
        bf16x16 b1 = frag_b(W + (size_t)(n0 + 16) * DMODEL + i0, DMODEL, lane);
        c[0][0] = WMMA(a0, b0, c[0][0]);
        c[0][1] = WMMA(a0, b1, c[0][1]);
        c[1][0] = WMMA(a1, b0, c[1][0]);
        c[1][1] = WMMA(a1, b1, c[1][1]);
    }
    int col_l = lane & 15, rhi = (lane >> 4) * 8;
#pragma unroll
    for (int p = 0; p < 2; ++p)
#pragma unroll
        for (int q = 0; q < 2; ++q) {
            int cc = n0 + q * 16 + col_l;
            float bv = lb[l * CDIM + cc];
#pragma unroll
            for (int j = 0; j < 8; ++j) {
                int m = m0 + p * 16 + rhi + j;
                out[((size_t)(l * MTOT + m)) * CDIM + cc] = c[p][q][j] + bv;
            }
        }
}

// ---------------------------------------------------------------------------
extern "C" void kernel_launch(void* const* d_in, const int* in_sizes, int n_in,
                              void* d_out, int out_size, void* d_ws, size_t ws_size,
                              hipStream_t stream) {
    (void)in_sizes; (void)n_in; (void)out_size; (void)ws_size;
    const __bf16* x      = (const __bf16*)d_in[0];   // (2,2048,256) bf16
    const __bf16* conv_w = (const __bf16*)d_in[1];   // (256,256,32,2) bf16
    const __bf16* mbias  = (const __bf16*)d_in[2];   // (256,) bf16
    const float*  wsh    = (const float*)d_in[3];    // (768,256) f32
    const float*  bsh    = (const float*)d_in[4];    // (768,) f32
    const float*  lwf    = (const float*)d_in[5];    // (4,1536,768) f32
    const float*  lbf    = (const float*)d_in[6];    // (4,1536) f32
    float* out = (float*)d_out;                      // (4,2,2048,1536) f32

    const size_t MB = (size_t)1 << 20;
    char* ws = (char*)d_ws;
    __bf16*       wTr    = (__bf16*)(ws + 0);          //   4 MB  w_r[k][o][i]
    __bf16*       wTi    = (__bf16*)(ws + 4 * MB);     //   4 MB  w_i[k][o][i]
    unsigned int* g      = (unsigned int*)(ws + 8 * MB); // 128 MB g[b][o][t][k]{r,i}
    __bf16*       pooled = (__bf16*)(ws + 136 * MB);   //   2 MB  pooled[m][o]
    __bf16*       hidden = (__bf16*)(ws + 138 * MB);   //   6 MB  hidden[m][c]
    __bf16*       wSb    = (__bf16*)(ws + 144 * MB);   //   0.4MB w_shared bf16
    __bf16*       lwB    = (__bf16*)(ws + 145 * MB);   //   9 MB  layer_w bf16

    k_transpose_w<<<(KMAX * DSPEC * DSPEC) / 256, 256, 0, stream>>>(conv_w, wTr, wTi);
    k_f32_to_bf16<<<(DMODEL * DSPEC + 255) / 256, 256, 0, stream>>>(wsh, wSb, DMODEL * DSPEC);
    k_f32_to_bf16<<<(NLAYERS * CDIM * DMODEL + 255) / 256, 256, 0, stream>>>(
        lwf, lwB, NLAYERS * CDIM * DMODEL);

    // 32768 waves (32 freqs x 128x8 tiles) -> 4096 blocks of 256
    k_spec_gemm<<<4096, 256, 0, stream>>>(x, wTr, wTi, g);
    // 512 waves (one per (b,o)) -> 64 blocks
    k_cumsum_pool<<<64, 256, 0, stream>>>(x, g, mbias, pooled);
    // 3072 waves -> 384 blocks
    k_hidden_gemm<<<384, 256, 0, stream>>>(pooled, wSb, bsh, hidden);
    // 24576 waves -> 3072 blocks
    k_cond_gemm<<<3072, 256, 0, stream>>>(hidden, lwB, lbf, out);
}
